// SpikeMLP_20151986552936
// MI455X (gfx1250) — matrix operand order repaired and checked by reading, not compiled, not run on hardware
//
#include <hip/hip_runtime.h>

typedef _Float16 v8h  __attribute__((ext_vector_type(8)));
typedef _Float16 v16h __attribute__((ext_vector_type(16)));
typedef float    v8f  __attribute__((ext_vector_type(8)));
typedef int      v4i_g __attribute__((vector_size(16)));   // matches builtin param

#define TSTEPS 16
#define BM  128      // block tile M
#define BNB 128      // block tile N
#define BK  32       // K step (one WMMA depth)
// wave tile: 32 (M) x 64 (N)  -> 2x4 accumulators, 8 waves (4m x 2n) per block

// ---- CDNA5 async global->LDS path (guarded; falls back to direct copy) ----
#if defined(__gfx1250__) && __has_builtin(__builtin_amdgcn_global_load_async_to_lds_b128)
#define HAVE_ASYNC 1
#else
#define HAVE_ASYNC 0
#endif

__device__ __forceinline__ void gl_async_b128(const _Float16* g, _Float16* l) {
#if HAVE_ASYNC
    __builtin_amdgcn_global_load_async_to_lds_b128(
        (__attribute__((address_space(1))) v4i_g*)(unsigned long long)g,
        (__attribute__((address_space(3))) v4i_g*)(unsigned)(unsigned long long)l,
        0, 0);
#else
    *(v8h*)l = *(const v8h*)g;
#endif
}

__device__ __forceinline__ void wait_async() {
#if HAVE_ASYNC
#if __has_builtin(__builtin_amdgcn_s_wait_asynccnt)
    __builtin_amdgcn_s_wait_asynccnt(0);
#else
    asm volatile("s_wait_asynccnt 0x0" ::: "memory");
#endif
#endif
}

// ---------------------------------------------------------------------------
// fp32 -> f16 weight cast (layout preserved: W is [N, K] row-major)
// ---------------------------------------------------------------------------
__global__ __launch_bounds__(256) void cast_w_f16(const float* __restrict__ src,
                                                  _Float16* __restrict__ dst, int n) {
    int i = blockIdx.x * 256 + threadIdx.x;
    if (i < n) dst[i] = (_Float16)src[i];
}

// ---------------------------------------------------------------------------
// spikes [B, I, T] fp32 -> A0 [B*T, I] f16   (m = b*16 + t)
// ---------------------------------------------------------------------------
__global__ __launch_bounds__(256) void pack_input_f16(const float* __restrict__ spk,
                                                      _Float16* __restrict__ A0,
                                                      int I, int n) {
    int idx = blockIdx.x * 256 + threadIdx.x;
    if (idx >= n) return;
    int i = idx % I;
    int m = idx / I;
    int b = m >> 4, t = m & 15;
    A0[idx] = (_Float16)spk[((size_t)b * I + i) * TSTEPS + t];
}

// ---------------------------------------------------------------------------
// out[b, f] = mean over t of spikes A[(b*16+t), f]   (f16 0/1 -> exact)
// ---------------------------------------------------------------------------
__global__ __launch_bounds__(256) void avg_spikes(const _Float16* __restrict__ A,
                                                  float* __restrict__ out,
                                                  int N, int n) {
    int idx = blockIdx.x * 256 + threadIdx.x;
    if (idx >= n) return;
    int f = idx % N, b = idx / N;
    float s = 0.f;
#pragma unroll
    for (int t = 0; t < TSTEPS; ++t)
        s += (float)A[(size_t)(b * TSTEPS + t) * N + f];
    out[idx] = s * (1.0f / TSTEPS);
}

// ---------------------------------------------------------------------------
// Fused layer: C = A[M,K] x W[N,K]^T (v_wmma_f32_16x16x32_f16), bias+BN affine,
// 16-step LIF scan, spikes written f16 into Anext[M,N] (next layer's A matrix).
//
// - M = B*T = 2048; T == 16 so each 16-row M stripe is one batch's timesteps.
// - A/B tiles double-buffered in LDS, streamed with GLOBAL_LOAD_ASYNC_TO_LDS_B128
//   (ASYNCcnt) so tile kt+1 loads while tile kt is multiplied.
// - LIF scan done in-registers: C-tile rows map t = r + 8*grp, the half-time
//   carry (volt) hops lanes n <-> n+16 with one ds_swizzle SWAPX16.
// ---------------------------------------------------------------------------
__global__ __launch_bounds__(256) void spike_layer(
    const _Float16* __restrict__ A,   // [M, K] f16
    const _Float16* __restrict__ W,   // [N, K] f16 (row-major original W[o,i])
    const float* __restrict__ bias,   // [N]
    const float* __restrict__ gamma,  // [N]
    const float* __restrict__ beta,   // [N]
    const float* __restrict__ mmean,  // [N]
    const float* __restrict__ mvar,   // [N]
    _Float16* __restrict__ Anext,     // [M, N] f16 spikes out
    int K, int N)
{
    __shared__ __align__(16) _Float16 sA[2][BM][BK + 8];
    __shared__ __align__(16) _Float16 sB[2][BNB][BK + 8];

    const int tid  = threadIdx.x;
    const int lane = tid & 31;
    const int wave = tid >> 5;
    const int wy   = wave >> 1;              // 0..3 : M sub-tile (32 rows)
    const int wx   = wave & 1;               // 0..1 : N sub-tile (64 cols)
    const int m0   = blockIdx.y * BM;
    const int n0   = blockIdx.x * BNB;
    const int grp  = (lane < 16) ? 0 : 1;    // lane half-group (ISA layout)
    const int lm   = lane & 15;

    v8f c[2][4] = {};                        // f32 accumulators (32x64)

    const int kTiles = K / BK;

    // ---- issue first tile (A: 512 chunks, B: 512 chunks; 4 asyncs/thread)
    {
#pragma unroll
        for (int r = 0; r < 2; ++r) {
            int chunk = tid + r * 256;           // 0..511
            int row = chunk >> 2;
            int cc  = (chunk & 3) * 8;
            gl_async_b128(&A[(size_t)(m0 + row) * K + cc], &sA[0][row][cc]);
            gl_async_b128(&W[(size_t)(n0 + row) * K + cc], &sB[0][row][cc]);
        }
    }

    for (int kt = 0; kt < kTiles; ++kt) {
        wait_async();          // own copies for tile kt complete
        __syncthreads();       // everyone's copies visible; prev compute done

        // ---- stream next tile into the other buffer while computing
        if (kt + 1 < kTiles) {
            const int kb = (kt + 1) * BK;
            const int nb = (kt + 1) & 1;
#pragma unroll
            for (int r = 0; r < 2; ++r) {
                int chunk = tid + r * 256;
                int row = chunk >> 2;
                int cc  = (chunk & 3) * 8;
                gl_async_b128(&A[(size_t)(m0 + row) * K + kb + cc], &sA[nb][row][cc]);
                gl_async_b128(&W[(size_t)(n0 + row) * K + kb + cc], &sB[nb][row][cc]);
            }
        }

        const int buf = kt & 1;

        // ---- register fragments per ISA 7.12.2 16-bit layouts
        v16h afrag[2], bfrag[4];
#pragma unroll
        for (int i = 0; i < 2; ++i) {
            const _Float16* pa = &sA[buf][wy * 32 + i * 16 + lm][0];
            v8h lo = *(const v8h*)(pa + (grp ? 8  : 0));   // K 0..7  / 8..15
            v8h hi = *(const v8h*)(pa + (grp ? 24 : 16));  // K 16..23 / 24..31
#pragma unroll
            for (int e = 0; e < 8; ++e) { afrag[i][e] = lo[e]; afrag[i][e + 8] = hi[e]; }
        }
#pragma unroll
        for (int j = 0; j < 4; ++j) {
            const _Float16* pb = &sB[buf][wx * 64 + j * 16 + lm][0];
            int ko = grp ? 8 : 0;                           // K 0..7,16..23 / 8..15,24..31
            v8h lo = *(const v8h*)(pb + ko);
            v8h hi = *(const v8h*)(pb + ko + 16);
#pragma unroll
            for (int e = 0; e < 8; ++e) { bfrag[j][e] = lo[e]; bfrag[j][e + 8] = hi[e]; }
        }

        // ---- 8x v_wmma_f32_16x16x32_f16
#pragma unroll
        for (int i = 0; i < 2; ++i)
#pragma unroll
            for (int j = 0; j < 4; ++j)
                c[i][j] = __builtin_amdgcn_wmma_f32_16x16x32_f16(
                    false, afrag[i], false, bfrag[j],
                    (short)0, c[i][j], false, false);
    }

    // ---- epilogue: bias + BN-inference affine coefficients (per column)
    float sc[4], sh[4];
#pragma unroll
    for (int j = 0; j < 4; ++j) {
        int n = n0 + wx * 64 + j * 16 + lm;
        sc[j] = gamma[n] / sqrtf(mvar[n] + 1e-5f);
        sh[j] = (bias[n] - mmean[n]) * sc[j] + beta[n];
    }

    // ---- LIF scan, all in registers. For each 16x16 C-tile: rows are
    // t = r + 8*grp. Lanes n (t=0..7) and n+16 (t=8..15) form one scan;
    // the carry is volt alone (spike = volt > VTH is a function of volt).
#pragma unroll
    for (int i = 0; i < 2; ++i) {
        int mb = m0 + wy * 32 + i * 16;                 // = batch * 16
#pragma unroll
        for (int j = 0; j < 4; ++j) {
            int n = n0 + wx * 64 + j * 16 + lm;
            // pass 1: lower half computes carry after t=7 (upper half: scratch)
            float volt = 0.f, sp = 0.f;
#pragma unroll
            for (int r = 0; r < 8; ++r) {
                float x = c[i][j][r] * sc[j] + sh[j];
                volt = volt * (1.0f - sp) * 0.75f + x;  // VDECAY, CDECAY=0
                sp   = (volt > 0.5f) ? 1.0f : 0.0f;     // VTH
            }
            // hop carry n <-> n+16 (ds_swizzle SWAPX16: xor=0x10, and=0x1f)
            float carry = __int_as_float(
                __builtin_amdgcn_ds_swizzle(__float_as_int(volt), 0x401F));
            volt = (lane < 16) ? 0.0f : carry;
            sp   = (volt > 0.5f) ? 1.0f : 0.0f;
            // pass 2: scan with correct initial state, emit spikes
#pragma unroll
            for (int r = 0; r < 8; ++r) {
                float x = c[i][j][r] * sc[j] + sh[j];
                volt = volt * (1.0f - sp) * 0.75f + x;
                sp   = (volt > 0.5f) ? 1.0f : 0.0f;
                Anext[(size_t)(mb + r + grp * 8) * N + n] = (_Float16)sp;
            }
        }
    }
}

// ---------------------------------------------------------------------------
extern "C" void kernel_launch(void* const* d_in, const int* in_sizes, int n_in,
                              void* d_out, int out_size, void* d_ws, size_t ws_size,
                              hipStream_t stream) {
    (void)in_sizes; (void)n_in; (void)out_size; (void)ws_size;
    const int B = 128, Din = 1024, H = 2048, Dout = 1024;
    const int M = B * TSTEPS;                       // 2048

    const float* spk = (const float*)d_in[0];
    const float* W0  = (const float*)d_in[1];
    const float* b0  = (const float*)d_in[2];
    const float* g0  = (const float*)d_in[3];
    const float* be0 = (const float*)d_in[4];
    const float* mm0 = (const float*)d_in[5];
    const float* mv0 = (const float*)d_in[6];
    const float* W1  = (const float*)d_in[7];
    const float* b1  = (const float*)d_in[8];
    const float* g1  = (const float*)d_in[9];
    const float* be1 = (const float*)d_in[10];
    const float* mm1 = (const float*)d_in[11];
    const float* mv1 = (const float*)d_in[12];
    const float* Wo  = (const float*)d_in[13];
    const float* bo  = (const float*)d_in[14];
    const float* go  = (const float*)d_in[15];
    const float* beo = (const float*)d_in[16];
    const float* mmo = (const float*)d_in[17];
    const float* mvo = (const float*)d_in[18];

    // workspace: two 8MB activation slots (ping-pong) + one 8MB weight slot
    char* ws = (char*)d_ws;
    _Float16* slotA = (_Float16*)(ws);                         // act ping
    _Float16* slotB = (_Float16*)(ws + (size_t)(8 << 20));     // act pong
    _Float16* wslot = (_Float16*)(ws + (size_t)(16 << 20));    // weights f16

    // input spikes -> A0 f16 [M, Din] in slotA
    {
        int n = M * Din;
        pack_input_f16<<<(n + 255) / 256, 256, 0, stream>>>(spk, slotA, Din, n);
    }
    // layer 0: [M,1024] x [2048,1024]^T -> spikes [M,2048] in slotB
    {
        int n = H * Din;
        cast_w_f16<<<(n + 255) / 256, 256, 0, stream>>>(W0, wslot, n);
        spike_layer<<<dim3(H / BNB, M / BM), 256, 0, stream>>>(
            slotA, wslot, b0, g0, be0, mm0, mv0, slotB, Din, H);
    }
    // layer 1: [M,2048] x [2048,2048]^T -> spikes [M,2048] in slotA
    {
        int n = H * H;
        cast_w_f16<<<(n + 255) / 256, 256, 0, stream>>>(W1, wslot, n);
        spike_layer<<<dim3(H / BNB, M / BM), 256, 0, stream>>>(
            slotB, wslot, b1, g1, be1, mm1, mv1, slotA, H, H);
    }
    // layer 2: [M,2048] x [1024,2048]^T -> spikes [M,1024] in slotB
    {
        int n = Dout * H;
        cast_w_f16<<<(n + 255) / 256, 256, 0, stream>>>(Wo, wslot, n);
        spike_layer<<<dim3(Dout / BNB, M / BM), 256, 0, stream>>>(
            slotA, wslot, bo, go, beo, mmo, mvo, slotB, H, Dout);
    }
    // average spikes over T -> fp32 output [B, Dout]
    {
        int n = B * Dout;
        avg_spikes<<<(n + 255) / 256, 256, 0, stream>>>(slotB, (float*)d_out, Dout, n);
    }
}
